// Block_28887950033544
// MI455X (gfx1250) — matrix operand order repaired and checked by reading, not compiled, not run on hardware
//
#include <hip/hip_runtime.h>
#include <math.h>

// ---------------- types ----------------
typedef __bf16 bf16;
typedef bf16 v16bf __attribute__((ext_vector_type(16)));
typedef bf16 v8bf  __attribute__((ext_vector_type(8)));
typedef float v8f  __attribute__((ext_vector_type(8)));

static __device__ inline v16bf cat8(v8bf lo, v8bf hi) {
  return __builtin_shufflevector(lo, hi, 0,1,2,3,4,5,6,7,8,9,10,11,12,13,14,15);
}

#define WMMA_BF16(a, b, c) \
  __builtin_amdgcn_wmma_f32_16x16x32_bf16(false, (a), false, (b), (short)0, (c), false, false)

// async 16-byte global -> LDS copy (ASYNCcnt); INST_OFFSET applies to both addresses.
static __device__ inline void async_b128(unsigned lds_off, const void* gptr) {
  unsigned long long ga = (unsigned long long)gptr;
  asm volatile("global_load_async_to_lds_b128 %0, %1, off"
               :: "v"(lds_off), "v"(ga) : "memory");
}
static __device__ inline void async_b128_off(unsigned lds_off, const void* gptr, int imm) {
  unsigned long long ga = (unsigned long long)gptr;
  if (imm == 16)
    asm volatile("global_load_async_to_lds_b128 %0, %1, off offset:16"
                 :: "v"(lds_off), "v"(ga) : "memory");
  else if (imm == 32)
    asm volatile("global_load_async_to_lds_b128 %0, %1, off offset:32"
                 :: "v"(lds_off), "v"(ga) : "memory");
  else
    asm volatile("global_load_async_to_lds_b128 %0, %1, off offset:48"
                 :: "v"(lds_off), "v"(ga) : "memory");
}
static __device__ inline void wait_async0() {
  asm volatile("s_wait_asynccnt 0" ::: "memory");
}
static __device__ inline unsigned lds_addr(const void* p) {
  return (unsigned)(unsigned long long)p;   // low 32 bits of generic addr = LDS offset
}

// ---------------- constants ----------------
static constexpr int Bsz = 2, T = 2048, C = 2048, H = 16, HS = 128;
static constexpr int Mtot = Bsz * T;          // 4096 rows
static constexpr int C3 = 3 * C;              // 6144
static constexpr float EPSF = 1e-5f;

// ---------------- cast fp32 -> bf16 ----------------
__global__ void cast_f32_bf16(const float* __restrict__ in, bf16* __restrict__ out, int n) {
  int i = blockIdx.x * blockDim.x + threadIdx.x;
  if (i < n) out[i] = (bf16)in[i];
}

// ---------------- layernorm: fp32 in -> bf16 out ----------------
__global__ __launch_bounds__(256)
void layernorm_bf16(const float* __restrict__ x, const float* __restrict__ w,
                    const float* __restrict__ b, bf16* __restrict__ out) {
  __shared__ float s1[256], s2[256];
  const int row = blockIdx.x;
  const int t = threadIdx.x;
  const float* xr = x + (size_t)row * C;
  float sum = 0.f, sq = 0.f;
  float v[8];
  const int base = t * 8;
#pragma unroll
  for (int i = 0; i < 8; ++i) { v[i] = xr[base + i]; sum += v[i]; sq += v[i] * v[i]; }
  s1[t] = sum; s2[t] = sq;
  __syncthreads();
  for (int off = 128; off > 0; off >>= 1) {
    if (t < off) { s1[t] += s1[t + off]; s2[t] += s2[t + off]; }
    __syncthreads();
  }
  const float mu = s1[0] * (1.0f / C);
  const float var = s2[0] * (1.0f / C) - mu * mu;
  const float rstd = rsqrtf(var + EPSF);
  bf16* orow = out + (size_t)row * C;
#pragma unroll
  for (int i = 0; i < 8; ++i) {
    const int c = base + i;
    orow[c] = (bf16)((v[i] - mu) * rstd * w[c] + b[c]);
  }
}

// ---------------- tiled WMMA bf16 GEMM (double-buffered, async A staging) -------
// out = epilogue(A[M,K] @ Bm[K,N] + bias)
// EPI 0: store bf16   EPI 1: gelu -> bf16   EPI 2: +resid -> fp32
template <int EPI>
__global__ __launch_bounds__(256)
void gemm_wmma(const bf16* __restrict__ A, const bf16* __restrict__ Bm,
               const float* __restrict__ bias, const float* __restrict__ resid,
               void* __restrict__ outp, int M, int N, int K) {
  __shared__ bf16 ldsA[2][128 * 32];   // row-major [row][k]
  __shared__ bf16 ldsBT[2][128 * 32];  // transposed [n][k]

  const int t = threadIdx.x;
  const int lane = t & 31, wave = t >> 5;
  const int half = lane >> 4, l16 = lane & 15;
  const int bx = blockIdx.x, by = blockIdx.y;
  const int wm = (wave >> 2) * 64, wn = (wave & 3) * 32;

  v8f acc[4][2];
#pragma unroll
  for (int mt = 0; mt < 4; ++mt)
#pragma unroll
    for (int nt = 0; nt < 2; ++nt) acc[mt][nt] = {};

  const int aRow = t >> 1, aCol = (t & 1) * 16;      // A tile: 128x32, 32B/thread
  const int bK = t >> 3, bN0 = (t & 7) * 16;          // B tile: 32x128

  auto stage = [&](int bk, int buf) {
    // A: two async 16B copies straight into LDS
    const bf16* ag = A + (size_t)(by * 128 + aRow) * K + bk + aCol;
    unsigned la = lds_addr(&ldsA[buf][aRow * 32 + aCol]);
    async_b128(la, ag);
    async_b128_off(la, ag, 16);
    // B: load row-major, transpose-scatter to [n][k]
    const bf16* bg = Bm + (size_t)(bk + bK) * N + bx * 128 + bN0;
    v16bf bv = *(const v16bf*)bg;
#pragma unroll
    for (int i = 0; i < 16; ++i) ldsBT[buf][(bN0 + i) * 32 + bK] = bv[i];
    // prefetch two K-tiles ahead into L2 (speculative; dropped if OOB)
    if (bk + 64 < K) {
      __builtin_prefetch(ag + 64, 0, 1);
      __builtin_prefetch(bg + (size_t)64 * N, 0, 1);
    }
  };

  const int nk = K / 32;
  stage(0, 0);
  for (int i = 0; i < nk; ++i) {
    const int buf = i & 1;
    wait_async0();          // my async copies for `buf` done
    __syncthreads();        // everyone's copies visible
    if (i + 1 < nk) stage((i + 1) * 32, buf ^ 1);

    v16bf af[4];
#pragma unroll
    for (int mt = 0; mt < 4; ++mt) {
      const bf16* p = &ldsA[buf][(wm + mt * 16 + l16) * 32 + half * 16];
      af[mt] = *(const v16bf*)p;
    }
#pragma unroll
    for (int nt = 0; nt < 2; ++nt) {
      const bf16* p = &ldsBT[buf][(wn + nt * 16 + l16) * 32 + half * 16];
      v16bf bfrag = *(const v16bf*)p;
#pragma unroll
      for (int mt = 0; mt < 4; ++mt)
        acc[mt][nt] = WMMA_BF16(af[mt], bfrag, acc[mt][nt]);
    }
    __syncthreads();        // done reading `buf` before it is restaged
  }

  // epilogue: C layout VGPR r -> row = r + 8*halflane, col = l16
#pragma unroll
  for (int mt = 0; mt < 4; ++mt)
#pragma unroll
    for (int nt = 0; nt < 2; ++nt)
#pragma unroll
      for (int r = 0; r < 8; ++r) {
        const int row = by * 128 + wm + mt * 16 + r + half * 8;
        const int col = bx * 128 + wn + nt * 16 + l16;
        const size_t idx = (size_t)row * N + col;
        float v = acc[mt][nt][r] + bias[col];
        if (EPI == 0) {
          ((bf16*)outp)[idx] = (bf16)v;
        } else if (EPI == 1) {
          v = 0.5f * v * (1.0f + erff(v * 0.70710678118654752f));
          ((bf16*)outp)[idx] = (bf16)v;
        } else {
          ((float*)outp)[idx] = v + resid[idx];
        }
      }
}

// ---------------- flash attention ----------------
// qkv bf16 [B*T][3C]; out bf16 [B*T][C]. One block per (b, h, 128-query tile).
__global__ __launch_bounds__(256)
void flash_attn(const bf16* __restrict__ qkv, bf16* __restrict__ out) {
  __shared__ bf16 ldsK[64 * 128];    // [key][d]
  __shared__ bf16 ldsVT[128 * 64];   // [d][key]
  __shared__ bf16 ldsP[8 * 16 * 64]; // per wave 16x64

  const int t = threadIdx.x;
  const int lane = t & 31, wave = t >> 5;
  const int half = lane >> 4, l16 = lane & 15;
  const int qt = blockIdx.x, h = blockIdx.y, b = blockIdx.z;
  const int q0 = qt * 128 + wave * 16;

  // Q fragments: A layout 16 x 128 -> 4 k-steps of 32
  v16bf qf[4];
  {
    const bf16* qp = qkv + (size_t)(b * T + q0 + l16) * C3 + h * HS;
#pragma unroll
    for (int kk = 0; kk < 4; ++kk) {
      const bf16* p = qp + kk * 32 + half * 16;
      qf[kk] = *(const v16bf*)p;
    }
  }

  v8f O[8];
#pragma unroll
  for (int ht = 0; ht < 8; ++ht) O[ht] = {};
  float mrow[8], lrow[8];
#pragma unroll
  for (int r = 0; r < 8; ++r) { mrow[r] = -1e30f; lrow[r] = 0.f; }

  const float scale = 0.088388347648318440f;  // 1/sqrt(128)
  const int nkt = 2 * qt + 2;                 // causal: key tiles of 64

  for (int kt = 0; kt < nkt; ++kt) {
    // stage K (async, row-major) and V (transposed via VGPRs) into LDS
    {
      const int key = t >> 2;             // 0..63
      const int d0 = (t & 3) * 32;        // 0,32,64,96
      const size_t rowbase = (size_t)(b * T + kt * 64 + key) * C3 + h * HS + d0;
      const bf16* kg = qkv + C + rowbase;
      unsigned lk = lds_addr(&ldsK[key * 128 + d0]);   // 64B per thread
      async_b128(lk, kg);
      async_b128_off(lk, kg, 16);
      async_b128_off(lk, kg, 32);
      async_b128_off(lk, kg, 48);
      const v16bf v0 = ((const v16bf*)(qkv + 2 * C + rowbase))[0];
      const v16bf v1 = ((const v16bf*)(qkv + 2 * C + rowbase))[1];
#pragma unroll
      for (int i = 0; i < 16; ++i) {
        ldsVT[(d0 + i) * 64 + key] = v0[i];
        ldsVT[(d0 + 16 + i) * 64 + key] = v1[i];
      }
    }
    wait_async0();
    __syncthreads();

    if (kt * 64 <= q0 + 15) {   // this wave has unmasked keys in the tile
      // S = Q @ K^T : 4 n-tiles of 16 keys
      v8f s[4];
#pragma unroll
      for (int nt = 0; nt < 4; ++nt) {
        s[nt] = {};
#pragma unroll
        for (int kk = 0; kk < 4; ++kk) {
          const bf16* p = ldsK + (nt * 16 + l16) * 128 + kk * 32 + half * 16;
          v8f tmpc = s[nt];
          s[nt] = WMMA_BF16(qf[kk], *(const v16bf*)p, tmpc);
        }
      }
      // scale + causal mask
#pragma unroll
      for (int nt = 0; nt < 4; ++nt)
#pragma unroll
        for (int r = 0; r < 8; ++r) {
          float v = s[nt][r] * scale;
          const int rg = q0 + r + half * 8;
          const int cg = kt * 64 + nt * 16 + l16;
          if (cg > rg) v = -1e30f;
          s[nt][r] = v;
        }
      // online softmax stats
      float alpha[8], mnew[8], rs[8];
#pragma unroll
      for (int r = 0; r < 8; ++r) {
        float tm = fmaxf(fmaxf(s[0][r], s[1][r]), fmaxf(s[2][r], s[3][r]));
#pragma unroll
        for (int off = 8; off >= 1; off >>= 1) tm = fmaxf(tm, __shfl_xor(tm, off));
        mnew[r] = fmaxf(mrow[r], tm);
        alpha[r] = __expf(mrow[r] - mnew[r]);
        mrow[r] = mnew[r];
        rs[r] = 0.f;
      }
#pragma unroll
      for (int nt = 0; nt < 4; ++nt)
#pragma unroll
        for (int r = 0; r < 8; ++r) {
          const float p = __expf(s[nt][r] - mnew[r]);
          s[nt][r] = p;
          rs[r] += p;
        }
#pragma unroll
      for (int r = 0; r < 8; ++r) {
#pragma unroll
        for (int off = 8; off >= 1; off >>= 1) rs[r] += __shfl_xor(rs[r], off);
        lrow[r] = lrow[r] * alpha[r] + rs[r];
#pragma unroll
        for (int ht = 0; ht < 8; ++ht) O[ht][r] *= alpha[r];
      }
      // P (C layout fp32) -> LDS bf16, reload as A fragments
      bf16* pw = ldsP + wave * 1024;
#pragma unroll
      for (int nt = 0; nt < 4; ++nt)
#pragma unroll
        for (int r = 0; r < 8; ++r)
          pw[(r + half * 8) * 64 + nt * 16 + l16] = (bf16)s[nt][r];
#pragma unroll
      for (int kk = 0; kk < 2; ++kk) {
        const bf16* p = pw + l16 * 64 + kk * 32 + half * 16;
        v16bf pf = *(const v16bf*)p;
#pragma unroll
        for (int ht = 0; ht < 8; ++ht) {
          const bf16* vp = ldsVT + (ht * 16 + l16) * 64 + kk * 32 + half * 16;
          O[ht] = WMMA_BF16(pf, *(const v16bf*)vp, O[ht]);
        }
      }
    }
    __syncthreads();
  }

  // write output (B,T,H,HS) flattened to [B*T][C]
#pragma unroll
  for (int ht = 0; ht < 8; ++ht)
#pragma unroll
    for (int r = 0; r < 8; ++r) {
      const int rg = q0 + r + half * 8;
      const int col = h * HS + ht * 16 + l16;
      out[(size_t)(b * T + rg) * C + col] = (bf16)(O[ht][r] / lrow[r]);
    }
}

// ---------------- host orchestration ----------------
extern "C" void kernel_launch(void* const* d_in, const int* in_sizes, int n_in,
                              void* d_out, int out_size, void* d_ws, size_t ws_size,
                              hipStream_t stream) {
  const float* x      = (const float*)d_in[0];
  const float* ln1_w  = (const float*)d_in[1];
  const float* ln1_b  = (const float*)d_in[2];
  const float* w_qkv  = (const float*)d_in[3];
  const float* b_qkv  = (const float*)d_in[4];
  const float* w_proj = (const float*)d_in[5];
  const float* b_proj = (const float*)d_in[6];
  const float* ln2_w  = (const float*)d_in[7];
  const float* ln2_b  = (const float*)d_in[8];
  const float* w_fc   = (const float*)d_in[9];
  const float* b_fc   = (const float*)d_in[10];
  const float* w_fc2  = (const float*)d_in[11];
  const float* b_fc2  = (const float*)d_in[12];
  float* out = (float*)d_out;

  char* ws = (char*)d_ws;
  size_t off = 0;
  auto alloc = [&](size_t bytes) { char* p = ws + off; off += (bytes + 255) & ~size_t(255); return p; };
  bf16* wqkv_bf = (bf16*)alloc((size_t)C * C3 * 2);
  bf16* wproj_bf = (bf16*)alloc((size_t)C * C * 2);
  bf16* wfc_bf  = (bf16*)alloc((size_t)C * 4 * C * 2);
  bf16* wfc2_bf = (bf16*)alloc((size_t)4 * C * C * 2);
  bf16* hln     = (bf16*)alloc((size_t)Mtot * C * 2);
  bf16* qkv_bf  = (bf16*)alloc((size_t)Mtot * C3 * 2);
  bf16* attn_bf = (bf16*)alloc((size_t)Mtot * C * 2);
  bf16* fcact_bf = (bf16*)alloc((size_t)Mtot * 4 * C * 2);

  // 1) weights -> bf16
  auto cast = [&](const float* src, bf16* dst, int n) {
    cast_f32_bf16<<<(n + 255) / 256, 256, 0, stream>>>(src, dst, n);
  };
  cast(w_qkv, wqkv_bf, C * C3);
  cast(w_proj, wproj_bf, C * C);
  cast(w_fc, wfc_bf, C * 4 * C);
  cast(w_fc2, wfc2_bf, 4 * C * C);

  // 2) LN1
  layernorm_bf16<<<Mtot, 256, 0, stream>>>(x, ln1_w, ln1_b, hln);

  // 3) QKV = hln @ w_qkv + b_qkv  (bf16 out)
  gemm_wmma<0><<<dim3(C3 / 128, Mtot / 128), 256, 0, stream>>>(
      hln, wqkv_bf, b_qkv, nullptr, qkv_bf, Mtot, C3, C);

  // 4) flash attention -> attn_bf
  flash_attn<<<dim3(T / 128, H, Bsz), 256, 0, stream>>>(qkv_bf, attn_bf);

  // 5) x1 = x + attn @ w_proj + b_proj  (fp32 into d_out)
  gemm_wmma<2><<<dim3(C / 128, Mtot / 128), 256, 0, stream>>>(
      attn_bf, wproj_bf, b_proj, x, out, Mtot, C, C);

  // 6) LN2 of x1
  layernorm_bf16<<<Mtot, 256, 0, stream>>>(out, ln2_w, ln2_b, hln);

  // 7) h = gelu(hln @ w_fc + b_fc)  (bf16)
  gemm_wmma<1><<<dim3(4 * C / 128, Mtot / 128), 256, 0, stream>>>(
      hln, wfc_bf, b_fc, nullptr, fcact_bf, Mtot, 4 * C, C);

  // 8) out = x1 + h @ w_fc2 + b_fc2  (fp32, resid = d_out itself)
  gemm_wmma<2><<<dim3(C / 128, Mtot / 128), 256, 0, stream>>>(
      fcact_bf, wfc2_bf, b_fc2, out, out, Mtot, C, 4 * C);

  (void)in_sizes; (void)n_in; (void)out_size; (void)ws_size;
}
